// _PerformerEncoderLayer_22522808500623
// MI455X (gfx1250) — matrix operand order repaired and checked by reading, not compiled, not run on hardware
//
#include <hip/hip_runtime.h>

// ---------------------------------------------------------------------------
// Performer encoder layer on gfx1250 (MI455X).
// Big GEMMs: 128x128 block tile, async global->LDS double-buffered staging
// (global_load_async_to_lds_b128 + s_wait_asynccnt), 64x64 per-wave WMMA tile.
// Batched/odd-shape GEMMs: direct-from-L2 32x32 per-wave WMMA tile.
// ---------------------------------------------------------------------------

typedef __bf16 bf16_t;
typedef __attribute__((ext_vector_type(16))) __bf16 v16bf;
typedef __attribute__((ext_vector_type(8)))  __bf16 v8bf;
typedef __attribute__((ext_vector_type(8)))  float   v8f;

#define BATCH_N 4
#define SEQ_N   2048
#define DMODEL  1024
#define NHEAD   16
#define HDIM    64
#define DFF_N   4096
#define NFEAT   532
#define NFPAD   544               // NFEAT padded to multiple of 16
#define NTOK    (BATCH_N*SEQ_N)   // 8192
#define FROW    (NHEAD*NFPAD)     // 8704

enum { GF_BIAS=1, GF_RELU=2, GF_BF16=4, GF_TRANS=8, GF_RES=16 };

// ---------------------------------------------------------------------------
// Large GEMM: C[M,N] = A[M,K] * B[N,K]^T, M%128==0, N%128==0, K%32==0.
// Block: 128 threads (4 waves, 2x2), per-wave 64x64, per-block 128x128.
// K-slabs (128x32 bf16 for A and B) staged via async global->LDS, 2 buffers.
// ---------------------------------------------------------------------------
#define LDSP 80   // bytes per 32-element bf16 LDS row (64B data + 16B pad)

__launch_bounds__(128)
__global__ void gemm_bf16_wmma_lds(const bf16_t* __restrict__ A, int lda,
                                   const bf16_t* __restrict__ B, int ldb,
                                   void* __restrict__ Cv, int ldc,
                                   const float* __restrict__ bias,
                                   const float* __restrict__ res,
                                   int K, int flags)
{
    __shared__ __align__(16) char shA[2][128*LDSP];
    __shared__ __align__(16) char shB[2][128*LDSP];

    const int mBase = blockIdx.y * 128;
    const int nBase = blockIdx.x * 128;
    const int lane = threadIdx.x & 31;
    const int wave = threadIdx.x >> 5;
    const int wm = (wave >> 1) * 64;
    const int wn = (wave &  1) * 64;
    const int hi = lane >> 4, lo16 = lane & 15;

    // stage one 128x32 slab of A and of B into LDS buffer `buf`
    auto stage = [&](int buf, int k0){
#pragma unroll
        for (int it = 0; it < 4; ++it){
            const int ch  = threadIdx.x + it*128;   // 512 chunks of 16B
            const int row = ch >> 2;
            const int c   = ch & 3;
            {
                const bf16_t* g = A + (long)(mBase+row)*lda + k0 + c*8;
                void* l = (void*)(shA[buf] + row*LDSP + c*16);
                asm volatile("global_load_async_to_lds_b128 %0, %1, off"
                             :: "v"((unsigned)(unsigned long)l),
                                "v"((unsigned long long)(unsigned long)g)
                             : "memory");
            }
            {
                const bf16_t* g = B + (long)(nBase+row)*ldb + k0 + c*8;
                void* l = (void*)(shB[buf] + row*LDSP + c*16);
                asm volatile("global_load_async_to_lds_b128 %0, %1, off"
                             :: "v"((unsigned)(unsigned long)l),
                                "v"((unsigned long long)(unsigned long)g)
                             : "memory");
            }
        }
    };

    const v8f zero = {0.f,0.f,0.f,0.f,0.f,0.f,0.f,0.f};
    v8f acc[4][4];
#pragma unroll
    for (int i=0;i<4;i++)
#pragma unroll
        for (int j=0;j<4;j++) acc[i][j] = zero;

    stage(0, 0);
    int buf = 0;
    for (int k0 = 0; k0 < K; k0 += 32){
        asm volatile("s_wait_asynccnt 0" ::: "memory");  // my slab landed in LDS
        __syncthreads();                                 // everyone's slab landed,
                                                         // and prev buffer fully read
        if (k0 + 32 < K) stage(buf ^ 1, k0 + 32);

        v16bf bfrag[4];
#pragma unroll
        for (int ni=0; ni<4; ni++){
            const char* p = shB[buf] + (wn + ni*16 + lo16)*LDSP + hi*32;
            bfrag[ni] = *(const v16bf*)p;
        }
#pragma unroll
        for (int mi=0; mi<4; mi++){
            union { v16bf v; v8bf h[2]; } af;
            const char* p = shA[buf] + (wm + mi*16 + lo16)*LDSP + hi*32;
            af.h[0] = *(const v8bf*)p;          // K = hi*16 .. hi*16+7
            af.h[1] = *(const v8bf*)(p + 16);   // K = hi*16+8 .. hi*16+15
#pragma unroll
            for (int ni=0; ni<4; ni++)
                acc[mi][ni] = __builtin_amdgcn_wmma_f32_16x16x32_bf16(
                    false, af.v, false, bfrag[ni],
                    (short)0, acc[mi][ni], false, false);
        }
        buf ^= 1;
    }

    // epilogue: D VGPR r holds (m = r + hi*8, n = lo16) of its 16x16 tile
#pragma unroll
    for (int mi=0; mi<4; mi++){
        const int mb = mBase + wm + mi*16 + hi*8;
#pragma unroll
        for (int ni=0; ni<4; ni++){
            const int nc = nBase + wn + ni*16 + lo16;
            const float bv = (flags & GF_BIAS) ? bias[nc] : 0.f;
#pragma unroll
            for (int r=0;r<8;r++){
                float val = acc[mi][ni][r] + bv;
                const long mr = mb + r;
                if (flags & GF_RES)  val += res[mr*ldc + nc];
                if (flags & GF_RELU) val = val > 0.f ? val : 0.f;
                if (flags & GF_BF16) ((bf16_t*)Cv)[mr*ldc + nc] = (bf16_t)val;
                else                 ((float*)Cv)[mr*ldc + nc]  = val;
            }
        }
    }
}

// ---------------------------------------------------------------------------
// Generic batched bf16 WMMA GEMM (direct from L2): C = A[M,K] * B[N,K]^T.
// batch z -> (bo=z/bdiv, bi=z%bdiv); wave computes 32x32; block 64x64.
// ---------------------------------------------------------------------------
__launch_bounds__(128)
__global__ void gemm_bf16_wmma(const bf16_t* __restrict__ A, int lda, long aOuter, long aInner,
                               const bf16_t* __restrict__ B, int ldb, long bOuter, long bInner,
                               void* __restrict__ Cv, int ldc, long cOuter, long cInner,
                               const float* __restrict__ bias,
                               const float* __restrict__ res,
                               int M, int N, int K, int bdiv, int flags)
{
    const int z  = blockIdx.z;
    const int bo = z / bdiv, bi = z % bdiv;
    const bf16_t* Ab = A + bo*aOuter + bi*aInner;
    const bf16_t* Bb = B + bo*bOuter + bi*bInner;
    const long  cOff = bo*cOuter + bi*cInner;

    const int lane = threadIdx.x & 31;
    const int wave = threadIdx.x >> 5;
    const int m0 = blockIdx.y*64 + (wave>>1)*32;
    const int n0 = blockIdx.x*64 + (wave&1)*32;

    const int hi    = lane >> 4;
    const int lo16  = lane & 15;
    const int koffA = hi*16;
    const int koffB = hi*16;

    bool mv[2], nv[2];
    int  arow[2], bcol[2];
#pragma unroll
    for (int t=0;t<2;t++){
        mv[t]   = (m0 + t*16 + 16) <= M;
        nv[t]   = (n0 + t*16 + 16) <= N;
        arow[t] = mv[t] ? (m0 + t*16 + lo16) : lo16;
        bcol[t] = nv[t] ? (n0 + t*16 + lo16) : lo16;
    }

    const v8f zero = {0.f,0.f,0.f,0.f,0.f,0.f,0.f,0.f};
    v8f acc[2][2] = {{zero,zero},{zero,zero}};

    union Afrag { v16bf v; v8bf h[2]; } af[2];

    for (int k0 = 0; k0 < K; k0 += 32) {
#pragma unroll
        for (int t=0;t<2;t++){
            const bf16_t* p = Ab + (long)arow[t]*lda + (k0 + koffA);
            af[t].h[0] = *(const v8bf*)p;
            af[t].h[1] = *(const v8bf*)(p + 8);
            if (k0 + 32 < K) __builtin_prefetch((const void*)(p + 32), 0, 1);
        }
        v16bf bfv[2];
#pragma unroll
        for (int t=0;t<2;t++){
            bfv[t] = *(const v16bf*)(Bb + (long)bcol[t]*ldb + (k0 + koffB));
        }
#pragma unroll
        for (int a=0;a<2;a++)
#pragma unroll
            for (int b2=0;b2<2;b2++)
                acc[a][b2] = __builtin_amdgcn_wmma_f32_16x16x32_bf16(
                    false, af[a].v, false, bfv[b2],
                    (short)0, acc[a][b2], false, false);
    }

#pragma unroll
    for (int a=0;a<2;a++){
        if (!mv[a]) continue;
#pragma unroll
        for (int b2=0;b2<2;b2++){
            if (!nv[b2]) continue;
            const int ncol  = n0 + b2*16 + lo16;
            const int mbase = m0 + a*16 + hi*8;
#pragma unroll
            for (int r=0;r<8;r++){
                float val = acc[a][b2][r];
                const int mr = mbase + r;
                if (flags & GF_BIAS) val += bias[ncol];
                if (flags & GF_RES)  val += res[cOff + (long)mr*ldc + ncol];
                if (flags & GF_RELU) val = val > 0.f ? val : 0.f;
                const long idx = (flags & GF_TRANS)
                    ? (cOff + (long)ncol*ldc + mr)
                    : (cOff + (long)mr*ldc + ncol);
                if (flags & GF_BF16) ((bf16_t*)Cv)[idx] = (bf16_t)val;
                else                 ((float*)Cv)[idx]  = val;
            }
        }
    }
}

// ---------------------------------------------------------------------------
// Elementwise / reduction kernels (fp32)
// ---------------------------------------------------------------------------
__global__ void k_f32_to_bf16(const float* __restrict__ s, bf16_t* __restrict__ d, long n){
    long i = (long)blockIdx.x*blockDim.x + threadIdx.x;
    if (i < n) d[i] = (bf16_t)s[i];
}

__global__ void k_pack_proj(const float* __restrict__ p, bf16_t* __restrict__ d){
    int i = blockIdx.x*blockDim.x + threadIdx.x;     // over NFPAD*HDIM
    if (i >= NFPAD*HDIM) return;
    int f = i / HDIM;
    d[i] = (f < NFEAT) ? (bf16_t)p[i] : (bf16_t)0.f;
}

__global__ void k_pack_bias3(const float* __restrict__ a, const float* __restrict__ b,
                             const float* __restrict__ c, float* __restrict__ d){
    int i = blockIdx.x*blockDim.x + threadIdx.x;
    if (i < DMODEL){ d[i]=a[i]; d[i+DMODEL]=b[i]; d[i+2*DMODEL]=c[i]; }
}

__launch_bounds__(256)
__global__ void k_layernorm(const float* __restrict__ x, const float* __restrict__ g,
                            const float* __restrict__ b, bf16_t* __restrict__ o){
    __shared__ float red[256];
    const int row = blockIdx.x;
    const float* xr = x + (long)row*DMODEL;
    float s = 0.f;
    for (int i=threadIdx.x; i<DMODEL; i+=256) s += xr[i];
    red[threadIdx.x]=s; __syncthreads();
    for (int off=128; off>0; off>>=1){
        if (threadIdx.x<off) red[threadIdx.x]+=red[threadIdx.x+off];
        __syncthreads();
    }
    const float mu = red[0]*(1.f/DMODEL);
    __syncthreads();
    float v=0.f;
    for (int i=threadIdx.x; i<DMODEL; i+=256){ float t=xr[i]-mu; v+=t*t; }
    red[threadIdx.x]=v; __syncthreads();
    for (int off=128; off>0; off>>=1){
        if (threadIdx.x<off) red[threadIdx.x]+=red[threadIdx.x+off];
        __syncthreads();
    }
    const float rstd = rsqrtf(red[0]*(1.f/DMODEL) + 1e-5f);
    for (int i=threadIdx.x; i<DMODEL; i+=256)
        o[(long)row*DMODEL + i] = (bf16_t)((xr[i]-mu)*rstd*g[i] + b[i]);
}

// dn-scale q,k to bf16 and compute diag = 0.5*sum((dn*x)^2) per (token, head)
__global__ void k_prep_qk(const float* __restrict__ qkv, bf16_t* __restrict__ qs,
                          bf16_t* __restrict__ ks, float* __restrict__ dq, float* __restrict__ dk){
    int t = blockIdx.x*blockDim.x + threadIdx.x;
    if (t >= NTOK*NHEAD) return;
    int n = t / NHEAD, h = t % NHEAD;
    const float dn = 0.35355339059327373f;   // HDIM^{-1/4}
    const float* q = qkv + (long)n*3*DMODEL + h*HDIM;
    const float* k = q + DMODEL;
    float sq=0.f, sk=0.f;
    for (int d=0; d<HDIM; d++){
        float a = dn*q[d], b = dn*k[d];
        qs[(long)n*DMODEL + h*HDIM + d] = (bf16_t)a;
        ks[(long)n*DMODEL + h*HDIM + d] = (bf16_t)b;
        sq += a*a; sk += b*b;
    }
    dq[t] = 0.5f*sq; dk[t] = 0.5f*sk;
}

// v -> vT[b,h,d,l] bf16 (K-contiguous operand for the kvs GEMM)
__global__ void k_vt(const float* __restrict__ qkv, bf16_t* __restrict__ vt){
    long i = (long)blockIdx.x*blockDim.x + threadIdx.x;
    if (i >= (long)NTOK*DMODEL) return;
    int n = (int)(i / DMODEL), c = (int)(i % DMODEL);
    int h = c / HDIM, d = c % HDIM;
    int b = n / SEQ_N, l = n % SEQ_N;
    vt[(((long)(b*NHEAD + h)*HDIM + d)*SEQ_N) + l] =
        (bf16_t)qkv[(long)n*3*DMODEL + 2*DMODEL + c];
}

__global__ void k_stab_q(const float* __restrict__ dd, float* __restrict__ st){
    int t = blockIdx.x*blockDim.x + threadIdx.x;
    if (t >= NTOK*NHEAD) return;
    int n = t / NHEAD, h = t % NHEAD;
    const float* p = dd + (long)n*FROW + h*NFPAD;
    float m = -3.4e38f;
    for (int f=0; f<NFEAT; f++) m = fmaxf(m, p[f]);
    st[t] = m;
}

__launch_bounds__(256)
__global__ void k_stab_k(const float* __restrict__ dd, float* __restrict__ st){
    __shared__ float red[256];
    int b = blockIdx.x / NHEAD, h = blockIdx.x % NHEAD;
    float m = -3.4e38f;
    for (long i = threadIdx.x; i < (long)SEQ_N*NFEAT; i += 256){
        int l = (int)(i / NFEAT), f = (int)(i % NFEAT);
        m = fmaxf(m, dd[(long)(b*SEQ_N + l)*FROW + h*NFPAD + f]);
    }
    red[threadIdx.x]=m; __syncthreads();
    for (int off=128; off>0; off>>=1){
        if (threadIdx.x<off) red[threadIdx.x]=fmaxf(red[threadIdx.x],red[threadIdx.x+off]);
        __syncthreads();
    }
    if (threadIdx.x==0) st[blockIdx.x]=red[0];
}

__global__ void k_exp_q(const float* __restrict__ dd, const float* __restrict__ st,
                        const float* __restrict__ dg, bf16_t* __restrict__ qp){
    long i = (long)blockIdx.x*blockDim.x + threadIdx.x;
    if (i >= (long)NTOK*FROW) return;
    int n = (int)(i / FROW), c = (int)(i % FROW);
    int h = c / NFPAD, f = c % NFPAD;
    float v = 0.f;
    if (f < NFEAT){
        int t = n*NHEAD + h;
        v = 0.0433555051f * (__expf(dd[i] - st[t] - dg[t]) + 1e-6f);  // ratio=1/sqrt(532)
    }
    qp[i] = (bf16_t)v;
}

// exp for K path, masked, written transposed as kpT[b,h,m,l] bf16
__global__ void k_exp_k(const float* __restrict__ dd, const float* __restrict__ st,
                        const float* __restrict__ dg, const unsigned char* __restrict__ pad,
                        bf16_t* __restrict__ kpt){
    long i = (long)blockIdx.x*blockDim.x + threadIdx.x;
    if (i >= (long)NTOK*FROW) return;
    int n = (int)(i / FROW), c = (int)(i % FROW);
    int h = c / NFPAD, f = c % NFPAD;
    int b = n / SEQ_N, l = n % SEQ_N;
    float v = 0.f;
    if (f < NFEAT && pad[n] == 0){
        v = 0.0433555051f * (__expf(dd[i] - st[b*NHEAD + h] - dg[n*NHEAD + h]) + 1e-6f);
    }
    kpt[((long)(b*NHEAD+h)*NFPAD + f)*SEQ_N + l] = (bf16_t)v;
}

__global__ void k_kssum(const bf16_t* __restrict__ kpt, float* __restrict__ kss){
    int r = blockIdx.x*blockDim.x + threadIdx.x;
    if (r >= BATCH_N*NHEAD*NFPAD) return;
    const bf16_t* p = kpt + (long)r*SEQ_N;
    float s=0.f;
    for (int l=0;l<SEQ_N;l++) s += (float)p[l];
    kss[r]=s;
}

__global__ void k_denom(const bf16_t* __restrict__ qp, const float* __restrict__ kss,
                        float* __restrict__ den){
    int t = blockIdx.x*blockDim.x + threadIdx.x;
    if (t >= NTOK*NHEAD) return;
    int n = t / NHEAD, h = t % NHEAD;
    int b = n / SEQ_N;
    const bf16_t* qr = qp  + (long)n*FROW + h*NFPAD;
    const float*  kr = kss + (long)(b*NHEAD+h)*NFPAD;
    float s=0.f;
    for (int m=0;m<NFPAD;m++) s += (float)qr[m]*kr[m];
    den[t] = s + 1e-6f;
}

__global__ void k_attn(const float* __restrict__ num, const float* __restrict__ den,
                       bf16_t* __restrict__ o){
    long i = (long)blockIdx.x*blockDim.x + threadIdx.x;
    if (i >= (long)NTOK*DMODEL) return;
    int n = (int)(i / DMODEL), h = (int)((i % DMODEL)/HDIM);
    o[i] = (bf16_t)(num[i] / den[n*NHEAD + h]);
}

// ---------------------------------------------------------------------------
extern "C" void kernel_launch(void* const* d_in, const int* in_sizes, int n_in,
                              void* d_out, int out_size, void* d_ws, size_t ws_size,
                              hipStream_t stream)
{
    const float* src   = (const float*)d_in[0];
    const unsigned char* pad = (const unsigned char*)d_in[1];   // bool mask
    const float* proj  = (const float*)d_in[2];
    const float* q_w   = (const float*)d_in[3];
    const float* q_b   = (const float*)d_in[4];
    const float* k_w   = (const float*)d_in[5];
    const float* k_b   = (const float*)d_in[6];
    const float* v_w   = (const float*)d_in[7];
    const float* v_b   = (const float*)d_in[8];
    const float* out_w = (const float*)d_in[9];
    const float* out_b = (const float*)d_in[10];
    const float* ln1_g = (const float*)d_in[11];
    const float* ln1_b = (const float*)d_in[12];
    const float* ln2_g = (const float*)d_in[13];
    const float* ln2_b = (const float*)d_in[14];
    const float* ff1_w = (const float*)d_in[15];
    const float* ff1_b = (const float*)d_in[16];
    const float* ff2_w = (const float*)d_in[17];
    const float* ff2_b = (const float*)d_in[18];
    (void)in_sizes; (void)n_in; (void)out_size; (void)ws_size;

    char* ws = (char*)d_ws;
    size_t off = 0;
    auto alloc = [&](size_t bytes)->char*{
        off = (off + 511) & ~(size_t)511;
        char* p = ws + off; off += bytes; return p;
    };
    bf16_t* wqkv = (bf16_t*)alloc((size_t)3*DMODEL*DMODEL*2);
    float*  qkvb = (float*) alloc((size_t)3*DMODEL*4);
    bf16_t* wout = (bf16_t*)alloc((size_t)DMODEL*DMODEL*2);
    bf16_t* wff1 = (bf16_t*)alloc((size_t)DFF_N*DMODEL*2);
    bf16_t* wff2 = (bf16_t*)alloc((size_t)DMODEL*DFF_N*2);
    bf16_t* projb= (bf16_t*)alloc((size_t)NFPAD*HDIM*2);
    bf16_t* xnorm= (bf16_t*)alloc((size_t)NTOK*DMODEL*2);
    bf16_t* hnorm= (bf16_t*)alloc((size_t)NTOK*DMODEL*2);
    float*  qkv  = (float*) alloc((size_t)NTOK*3*DMODEL*4);   // reused as ffh (bf16)
    bf16_t* ffh  = (bf16_t*)qkv;
    bf16_t* qsc  = (bf16_t*)alloc((size_t)NTOK*DMODEL*2);
    bf16_t* ksc  = (bf16_t*)alloc((size_t)NTOK*DMODEL*2);
    float*  diagq= (float*) alloc((size_t)NTOK*NHEAD*4);
    float*  diagk= (float*) alloc((size_t)NTOK*NHEAD*4);
    float*  stabq= (float*) alloc((size_t)NTOK*NHEAD*4);
    float*  stabk= (float*) alloc((size_t)BATCH_N*NHEAD*4);
    bf16_t* vt   = (bf16_t*)alloc((size_t)NTOK*DMODEL*2);
    float*  dd   = (float*) alloc((size_t)NTOK*FROW*4);       // shared: K-path then Q-path
    bf16_t* kpt  = (bf16_t*)alloc((size_t)BATCH_N*NHEAD*NFPAD*SEQ_N*2); // reused as qp
    bf16_t* qp   = kpt;                                       // same size (71.3M elems)
    bf16_t* kvst = (bf16_t*)alloc((size_t)BATCH_N*NHEAD*HDIM*NFPAD*2);
    float*  kss  = (float*) alloc((size_t)BATCH_N*NHEAD*NFPAD*4);
    float*  den  = (float*) alloc((size_t)NTOK*NHEAD*4);
    float*  numer= (float*) alloc((size_t)NTOK*DMODEL*4);
    bf16_t* attnb= (bf16_t*)alloc((size_t)NTOK*DMODEL*2);
    float*  xaft = (float*) alloc((size_t)NTOK*DMODEL*4);

    const int TB = 256;
    const dim3 blk(128);

    // ---- weight conversion / packing ----
    k_f32_to_bf16<<<4096, TB, 0, stream>>>(q_w,  wqkv,                1048576);
    k_f32_to_bf16<<<4096, TB, 0, stream>>>(k_w,  wqkv + 1048576,      1048576);
    k_f32_to_bf16<<<4096, TB, 0, stream>>>(v_w,  wqkv + 2*1048576,    1048576);
    k_f32_to_bf16<<<4096, TB, 0, stream>>>(out_w, wout,               1048576);
    k_f32_to_bf16<<<16384, TB, 0, stream>>>(ff1_w, wff1,              4194304);
    k_f32_to_bf16<<<16384, TB, 0, stream>>>(ff2_w, wff2,              4194304);
    k_pack_proj<<<(NFPAD*HDIM+TB-1)/TB, TB, 0, stream>>>(proj, projb);
    k_pack_bias3<<<(DMODEL+TB-1)/TB, TB, 0, stream>>>(q_b, k_b, v_b, qkvb);

    // ---- ln1 ----
    k_layernorm<<<NTOK, 256, 0, stream>>>(src, ln1_g, ln1_b, xnorm);

    // ---- fused QKV GEMM: [8192,1024] x [3072,1024]^T (async-LDS path) ----
    gemm_bf16_wmma_lds<<<dim3(3*DMODEL/128, NTOK/128), blk, 0, stream>>>(
        xnorm, DMODEL, wqkv, DMODEL, qkv, 3*DMODEL,
        qkvb, nullptr, DMODEL, GF_BIAS);

    // ---- scale q,k by dn -> bf16, diag; transpose v ----
    k_prep_qk<<<(NTOK*NHEAD+TB-1)/TB, TB, 0, stream>>>(qkv, qsc, ksc, diagq, diagk);
    k_vt<<<(int)(((long)NTOK*DMODEL+TB-1)/TB), TB, 0, stream>>>(qkv, vt);

    // ---- K-path features: per head, [8192,64] x [544,64]^T ----
    gemm_bf16_wmma<<<dim3((NFPAD+63)/64, NTOK/64, NHEAD), blk, 0, stream>>>(
        ksc, DMODEL, 0, HDIM, projb, HDIM, 0, 0, dd, FROW, 0, NFPAD,
        nullptr, nullptr, NTOK, NFPAD, HDIM, NHEAD, 0);
    k_stab_k<<<BATCH_N*NHEAD, 256, 0, stream>>>(dd, stabk);
    k_exp_k<<<(int)(((long)NTOK*FROW+TB-1)/TB), TB, 0, stream>>>(dd, stabk, diagk, pad, kpt);
    k_kssum<<<(BATCH_N*NHEAD*NFPAD+TB-1)/TB, TB, 0, stream>>>(kpt, kss);

    // ---- kvs: per (b,h), kpT[544,2048] x vT[64,2048]^T -> kvsT[64,544] bf16 ----
    gemm_bf16_wmma<<<dim3(1, (NFPAD+63)/64, BATCH_N*NHEAD), blk, 0, stream>>>(
        kpt,  SEQ_N, (long)NHEAD*NFPAD*SEQ_N, (long)NFPAD*SEQ_N,
        vt,   SEQ_N, (long)NHEAD*HDIM*SEQ_N,  (long)HDIM*SEQ_N,
        kvst, NFPAD, (long)NHEAD*HDIM*NFPAD,  (long)HDIM*NFPAD,
        nullptr, nullptr, NFPAD, HDIM, SEQ_N, NHEAD, GF_BF16|GF_TRANS);

    // ---- Q-path features (dd buffer reused) ----
    gemm_bf16_wmma<<<dim3((NFPAD+63)/64, NTOK/64, NHEAD), blk, 0, stream>>>(
        qsc, DMODEL, 0, HDIM, projb, HDIM, 0, 0, dd, FROW, 0, NFPAD,
        nullptr, nullptr, NTOK, NFPAD, HDIM, NHEAD, 0);
    k_stab_q<<<(NTOK*NHEAD+TB-1)/TB, TB, 0, stream>>>(dd, stabq);
    k_exp_q<<<(int)(((long)NTOK*FROW+TB-1)/TB), TB, 0, stream>>>(dd, stabq, diagq, qp);

    // ---- numer: per (b,h), qp[2048,544] x kvsT[64,544]^T ----
    gemm_bf16_wmma<<<dim3(1, SEQ_N/64, BATCH_N*NHEAD), blk, 0, stream>>>(
        qp,    FROW,  (long)SEQ_N*FROW,           (long)NFPAD,
        kvst,  NFPAD, (long)NHEAD*HDIM*NFPAD,     (long)HDIM*NFPAD,
        numer, DMODEL,(long)SEQ_N*DMODEL,         (long)HDIM,
        nullptr, nullptr, SEQ_N, HDIM, NFPAD, NHEAD, 0);

    k_denom<<<(NTOK*NHEAD+TB-1)/TB, TB, 0, stream>>>(qp, kss, den);
    k_attn<<<(int)(((long)NTOK*DMODEL+TB-1)/TB), TB, 0, stream>>>(numer, den, attnb);

    // ---- out-proj + residual(src) -> x_after (async-LDS path) ----
    gemm_bf16_wmma_lds<<<dim3(DMODEL/128, NTOK/128), blk, 0, stream>>>(
        attnb, DMODEL, wout, DMODEL, xaft, DMODEL,
        out_b, src, DMODEL, GF_BIAS|GF_RES);

    // ---- ln2 ----
    k_layernorm<<<NTOK, 256, 0, stream>>>(xaft, ln2_g, ln2_b, hnorm);

    // ---- ff1 (+bias, relu, store bf16 into reused qkv buffer) ----
    gemm_bf16_wmma_lds<<<dim3(DFF_N/128, NTOK/128), blk, 0, stream>>>(
        hnorm, DMODEL, wff1, DMODEL, ffh, DFF_N,
        ff1_b, nullptr, DMODEL, GF_BIAS|GF_RELU|GF_BF16);

    // ---- ff2 (+bias, residual x_after) -> d_out fp32 (async-LDS path) ----
    gemm_bf16_wmma_lds<<<dim3(DMODEL/128, NTOK/128), blk, 0, stream>>>(
        ffh, DFF_N, wff2, DFF_N, d_out, DMODEL,
        ff2_b, xaft, DFF_N, GF_BIAS|GF_RES);
}
